// FenrirFC_65034394796800
// MI455X (gfx1250) — compile-verified, matrix operand order checked by reading, not run on hardware
//
#include <hip/hip_runtime.h>
#include <hip/hip_fp16.h>

typedef __attribute__((ext_vector_type(16))) _Float16 v16h;
typedef __attribute__((ext_vector_type(8)))  float    v8f;
typedef __attribute__((ext_vector_type(4)))  float    v4f;   // native clang vector for nontemporal loads

#define T_STEPS 100
#define BATCH   256
#define TB_ROWS (T_STEPS * BATCH)   // 25600 rows of the hoisted GEMM
#define KDIM    3600
#define NOUT1   64
#define NCLS    10
#define KTILES  113                 // ceil(3600 / 32); last tile is the 16-wide tail

// ---------------- workspace layout ----------------
// [0, 8)                       : 2x uint  -> max|w1|, max|w2| (float bits)
// [256, 256 + PB_BYTES)        : packed qw1^T in WMMA B-operand layout (f16)
// [CUR1_OFF, +25600*64*4)      : cur1 f32
#define PB_HALVES ((size_t)KTILES * 4 * 32 * 16)     // 231424 halves
#define PB_BYTES  (PB_HALVES * 2)                    // 462848 B
#define CUR1_OFF  (256 + PB_BYTES)                   // 463104 (256-aligned)

__global__ void init_max_kernel(unsigned* maxv) {
    if (threadIdx.x < 2) maxv[threadIdx.x] = 0u;
}

__global__ void reduce_absmax_kernel(const float* __restrict__ src, int n,
                                     unsigned* __restrict__ dst) {
    __shared__ float red[256];
    float m = 0.0f;
    for (int i = blockIdx.x * blockDim.x + threadIdx.x; i < n;
         i += gridDim.x * blockDim.x)
        m = fmaxf(m, fabsf(src[i]));
    red[threadIdx.x] = m;
    __syncthreads();
    for (int s = 128; s > 0; s >>= 1) {
        if (threadIdx.x < s)
            red[threadIdx.x] = fmaxf(red[threadIdx.x], red[threadIdx.x + s]);
        __syncthreads();
    }
    if (threadIdx.x == 0) atomicMax(dst, __float_as_uint(red[0]));
}

// Quantize w1 (4-bit symmetric, STE) and pre-swizzle qw1^T (=B, K x N) into the
// documented 16-bit B-operand lane layout for v_wmma_f32_16x16x32_f16:
// lane L (hi = L>>4, n = ntile*16 + (L&15)); element e in 0..15 -> K = e + 8*((e>>3)+hi).
// Storage: pB[((kt*4 + nt)*32 + lane)*16 + e], zero-padded for K >= 3600.
__global__ void pack_b_kernel(const float* __restrict__ w1,
                              const unsigned* __restrict__ maxv,
                              _Float16* __restrict__ pB) {
    const int tid = blockIdx.x * blockDim.x + threadIdx.x;
    if (tid >= KTILES * 4 * 32) return;
    const int lane = tid & 31;
    const int nt   = (tid >> 5) & 3;
    const int kt   = tid >> 7;
    const float s     = __uint_as_float(maxv[0]) * (1.0f / 7.0f);
    const float inv_s = (s > 0.0f) ? 1.0f / s : 0.0f;
    const int n  = nt * 16 + (lane & 15);
    const int hi = lane >> 4;
    __align__(16) _Float16 out[16];
#pragma unroll
    for (int e = 0; e < 16; ++e) {
        const int k = kt * 32 + e + 8 * ((e >> 3) + hi);
        float v = 0.0f;
        if (k < KDIM) {
            const float w = w1[(size_t)n * KDIM + k];
            v = fminf(fmaxf(rintf(w * inv_s), -8.0f), 7.0f) * s;
        }
        out[e] = (_Float16)v;
    }
    _Float16* d = pB + (size_t)tid * 16;
    *(uint4*)d       = *(uint4*)&out[0];
    *(uint4*)(d + 8) = *(uint4*)&out[8];
}

// Hoisted layer-1 GEMM: cur1[25600,64] = X[25600,3600] @ qw1^T, f16 WMMA, f32 acc.
// 128 threads = 4 waves; block tile 128(M) x 64(N): each wave owns 2 M-tiles x 4
// N-tiles (8 accumulators), halving B-operand L2 traffic per output. K-step 32
// staged through LDS; 112 guard-free K iterations + one guarded 16-wide tail.
__global__ void __launch_bounds__(128) gemm1_kernel(const float* __restrict__ x,
                                                    const _Float16* __restrict__ pB,
                                                    float* __restrict__ cur1) {
    __shared__ _Float16 lA[128][40];  // 32 K-cols + pad (80B row stride: conflict-free)
    const int tid  = threadIdx.x;
    const int wave = tid >> 5;        // 0..3
    const int lane = tid & 31;
    const int hi   = lane >> 4;
    const int ln16 = lane & 15;
    const int rowbase = blockIdx.x * 128;

    v8f c00 = {}, c01 = {}, c02 = {}, c03 = {};   // M-tile 0 (rows wave*16 + ..)
    v8f c10 = {}, c11 = {}, c12 = {}, c13 = {};   // M-tile 1 (rows 64 + wave*16 + ..)

    const int srow = tid >> 3;        // 0..15
    const int scol = (tid & 7) * 4;   // 0..28 step 4

#define STAGE_ROW(j, KB, VALID)                                                  \
    {                                                                            \
        const int r  = srow + (j) * 16;                                          \
        __align__(8) _Float16 h[4];                                              \
        if (VALID) {                                                             \
            const v4f f = __builtin_nontemporal_load(                            \
                (const v4f*)(x + (size_t)(rowbase + r) * KDIM + (KB) + scol));   \
            h[0] = (_Float16)f.x; h[1] = (_Float16)f.y;                          \
            h[2] = (_Float16)f.z; h[3] = (_Float16)f.w;                          \
        } else {                                                                 \
            h[0] = h[1] = h[2] = h[3] = (_Float16)0.0f;                          \
        }                                                                        \
        *(uint2*)&lA[r][scol] = *(uint2*)h;                                      \
    }

#define COMPUTE_STEP(KT)                                                          \
    {                                                                             \
        const int arow0 = wave * 16 + ln16;                                       \
        const int arow1 = 64 + wave * 16 + ln16;                                  \
        v16h a0, a1;                                                              \
        *(uint4*)&a0         = *(const uint4*)&lA[arow0][8 * hi];                 \
        *(((uint4*)&a0) + 1) = *(const uint4*)&lA[arow0][16 + 8 * hi];            \
        *(uint4*)&a1         = *(const uint4*)&lA[arow1][8 * hi];                 \
        *(((uint4*)&a1) + 1) = *(const uint4*)&lA[arow1][16 + 8 * hi];            \
        const _Float16* bp = pB + (size_t)(KT) * 4 * 32 * 16;                     \
        v16h b0, b1, b2, b3;                                                      \
        const _Float16* p0 = bp + (size_t)(0 * 32 + lane) * 16;                   \
        const _Float16* p1 = bp + (size_t)(1 * 32 + lane) * 16;                   \
        const _Float16* p2 = bp + (size_t)(2 * 32 + lane) * 16;                   \
        const _Float16* p3 = bp + (size_t)(3 * 32 + lane) * 16;                   \
        *(uint4*)&b0 = *(const uint4*)p0; *(((uint4*)&b0)+1) = *(const uint4*)(p0+8); \
        *(uint4*)&b1 = *(const uint4*)p1; *(((uint4*)&b1)+1) = *(const uint4*)(p1+8); \
        *(uint4*)&b2 = *(const uint4*)p2; *(((uint4*)&b2)+1) = *(const uint4*)(p2+8); \
        *(uint4*)&b3 = *(const uint4*)p3; *(((uint4*)&b3)+1) = *(const uint4*)(p3+8); \
        c00 = __builtin_amdgcn_wmma_f32_16x16x32_f16(false, a0, false, b0, (short)0, c00, false, false); \
        c01 = __builtin_amdgcn_wmma_f32_16x16x32_f16(false, a0, false, b1, (short)0, c01, false, false); \
        c02 = __builtin_amdgcn_wmma_f32_16x16x32_f16(false, a0, false, b2, (short)0, c02, false, false); \
        c03 = __builtin_amdgcn_wmma_f32_16x16x32_f16(false, a0, false, b3, (short)0, c03, false, false); \
        c10 = __builtin_amdgcn_wmma_f32_16x16x32_f16(false, a1, false, b0, (short)0, c10, false, false); \
        c11 = __builtin_amdgcn_wmma_f32_16x16x32_f16(false, a1, false, b1, (short)0, c11, false, false); \
        c12 = __builtin_amdgcn_wmma_f32_16x16x32_f16(false, a1, false, b2, (short)0, c12, false, false); \
        c13 = __builtin_amdgcn_wmma_f32_16x16x32_f16(false, a1, false, b3, (short)0, c13, false, false); \
    }

    // ---- 112 guard-free K iterations (K = 0..3583) ----
    for (int kt = 0; kt < KTILES - 1; ++kt) {
        const int kb = kt * 32;
#pragma unroll
        for (int j = 0; j < 8; ++j) STAGE_ROW(j, kb, true)
        // prefetch next K tile of this thread's first row
        __builtin_prefetch(x + (size_t)(rowbase + srow) * KDIM + kb + 32 + scol, 0, 1);
        __syncthreads();
        COMPUTE_STEP(kt)
        __syncthreads();
    }

    // ---- guarded tail: kt = 112 covers K = 3584..3599 (cols >= 16 zeroed) ----
    {
        const bool valid = scol < 16;   // 3584 + scol + 3 < 3600
#pragma unroll
        for (int j = 0; j < 8; ++j) STAGE_ROW(j, (KTILES - 1) * 32, valid)
        __syncthreads();
        COMPUTE_STEP(KTILES - 1)
    }
#undef STAGE_ROW
#undef COMPUTE_STEP

    // ---- store: C/D layout -> VGPR r holds M = r + 8*hi, N = ln16 ----
#pragma unroll
    for (int r = 0; r < 8; ++r) {
        const int rg0 = rowbase + wave * 16 + r + 8 * hi;
        float* o0 = cur1 + (size_t)rg0 * NOUT1 + ln16;
        o0[0]  = c00[r];
        o0[16] = c01[r];
        o0[32] = c02[r];
        o0[48] = c03[r];
        float* o1 = o0 + (size_t)64 * NOUT1;
        o1[0]  = c10[r];
        o1[16] = c11[r];
        o1[32] = c12[r];
        o1[48] = c13[r];
    }
}

// Serial-in-T LIF scan. One wave per batch element; lane ln owns neurons ln and
// ln+32 of layer 1; mem2[0..9] replicated per-lane; 64->10 contraction via
// wave32 xor-shuffles (ds permute path).
__global__ void __launch_bounds__(32) lif_scan_kernel(
    const float* __restrict__ cur1, const float* __restrict__ w2,
    const unsigned* __restrict__ maxv,
    const float* __restrict__ pbeta1, const float* __restrict__ pbeta2,
    const float* __restrict__ pthr1,  const float* __restrict__ pthr2,
    float* __restrict__ out) {
    const int b  = blockIdx.x;   // 0..255
    const int ln = threadIdx.x;  // 0..31

    const float s1 = __uint_as_float(maxv[0]) * (1.0f / 7.0f);
    const float s2 = __uint_as_float(maxv[1]) * (1.0f / 7.0f);
    const float lim1 = s1 * 100.0f, lim2 = s2 * 100.0f;
    const float b1 = fminf(fmaxf(pbeta1[0], 0.0f), 1.0f);
    const float b2 = fminf(fmaxf(pbeta2[0], 0.0f), 1.0f);
    const float thr1 = pthr1[0], thr2 = pthr2[0];
    const float inv_s2 = (s2 > 0.0f) ? 1.0f / s2 : 0.0f;

    float w2q0[NCLS], w2q1[NCLS];
#pragma unroll
    for (int j = 0; j < NCLS; ++j) {
        const float wa = w2[j * NOUT1 + ln];
        const float wb = w2[j * NOUT1 + 32 + ln];
        w2q0[j] = fminf(fmaxf(rintf(wa * inv_s2), -8.0f), 7.0f) * s2;
        w2q1[j] = fminf(fmaxf(rintf(wb * inv_s2), -8.0f), 7.0f) * s2;
    }

    float m1a = 0.0f, m1b = 0.0f;
    float m2[NCLS];
#pragma unroll
    for (int j = 0; j < NCLS; ++j) m2[j] = 0.0f;

    for (int t = 0; t < T_STEPS; ++t) {
        const size_t base = ((size_t)t * BATCH + b) * NOUT1;
        const float ca = cur1[base + ln];
        const float cb = cur1[base + 32 + ln];
        m1a = fminf(fmaxf(m1a, -lim1), lim1) + ca;       // clamp then add
        m1b = fminf(fmaxf(m1b, -lim1), lim1) + cb;
        const float sa = (m1a - thr1) > 0.0f ? 1.0f : 0.0f;
        const float sb = (m1b - thr1) > 0.0f ? 1.0f : 0.0f;
        m1a = m1a * (1.0f - sa) * b1;                    // zero reset + decay
        m1b = m1b * (1.0f - sb) * b1;
#pragma unroll
        for (int j = 0; j < NCLS; ++j) {
            float p = sa * w2q0[j] + sb * w2q1[j];
            p += __shfl_xor(p, 16);
            p += __shfl_xor(p, 8);
            p += __shfl_xor(p, 4);
            p += __shfl_xor(p, 2);
            p += __shfl_xor(p, 1);
            float m = fminf(fmaxf(m2[j], -lim2), lim2) + p;
            const float sp = (m - thr2) > 0.0f ? 1.0f : 0.0f;
            m2[j] = m * (1.0f - sp) * b2;
            if (ln == 0)
                out[((size_t)t * BATCH + b) * NCLS + j] = sp;
        }
    }
}

extern "C" void kernel_launch(void* const* d_in, const int* in_sizes, int n_in,
                              void* d_out, int out_size, void* d_ws, size_t ws_size,
                              hipStream_t stream) {
    const float* x     = (const float*)d_in[0];  // [100,256,1,60,60]
    const float* w1    = (const float*)d_in[1];  // [64,3600]
    const float* w2    = (const float*)d_in[2];  // [10,64]
    const float* beta1 = (const float*)d_in[3];
    const float* beta2 = (const float*)d_in[4];
    const float* thr1  = (const float*)d_in[5];
    const float* thr2  = (const float*)d_in[6];
    float* out = (float*)d_out;                  // [100,256,10]

    unsigned char* ws = (unsigned char*)d_ws;
    unsigned*  maxv = (unsigned*)ws;
    _Float16*  pB   = (_Float16*)(ws + 256);
    float*     cur1 = (float*)(ws + CUR1_OFF);

    init_max_kernel<<<1, 32, 0, stream>>>(maxv);
    reduce_absmax_kernel<<<128, 256, 0, stream>>>(w1, NOUT1 * KDIM, maxv + 0);
    reduce_absmax_kernel<<<4, 256, 0, stream>>>(w2, NCLS * NOUT1, maxv + 1);
    pack_b_kernel<<<(KTILES * 4 * 32 + 255) / 256, 256, 0, stream>>>(w1, maxv, pB);
    gemm1_kernel<<<TB_ROWS / 128, 128, 0, stream>>>(x, pB, cur1);
    lif_scan_kernel<<<BATCH, 32, 0, stream>>>(cur1, w2, maxv, beta1, beta2,
                                              thr1, thr2, out);
}
